// CausalSelfAttention_6330781795003
// MI455X (gfx1250) — matrix operand order repaired and checked by reading, not compiled, not run on hardware
//
#include <hip/hip_runtime.h>

// Problem constants (match reference)
constexpr int kT = 2048;   // sequence length
constexpr int kC = 1024;   // channels
constexpr int kH = 16;     // heads
constexpr int kD = 64;     // head dim (kC / kH)

typedef __attribute__((ext_vector_type(16))) __bf16 v16bf;
typedef __attribute__((ext_vector_type(8)))  float  v8f;
typedef unsigned int uint32;

// fp32 -> bf16 round-to-nearest-even
static __device__ __forceinline__ unsigned short f2bf(float f) {
    uint32 u = __float_as_uint(f);
    u += 0x7FFFu + ((u >> 16) & 1u);
    return (unsigned short)(u >> 16);
}

// ---------------------------------------------------------------------------
// 1) elementwise fp32 -> bf16
__global__ void cvt_bf16_kernel(const float* __restrict__ src,
                                unsigned short* __restrict__ dst, int n) {
    int i = blockIdx.x * blockDim.x + threadIdx.x;
    if (i < n) dst[i] = f2bf(src[i]);
}

// ---------------------------------------------------------------------------
// 2) pack weight W (K x N, row-major f32) into WMMA B-fragment layout, bf16.
//    Packed index: (((nt*(K/32) + kt)*32 + lane)*16 + e)
//    holds B[k][n] with k = kt*32 + 16*(lane/16) + e, n = nt*16 + lane%16.
//    => each wave's B fragment for tile (nt,kt) is 32 contiguous bytes/lane.
__global__ void pack_w_kernel(const float* __restrict__ W,
                              unsigned short* __restrict__ out, int K, int N) {
    int i = blockIdx.x * blockDim.x + threadIdx.x;
    if (i >= K * N) return;
    int e    = i & 15;
    int lane = (i >> 4) & 31;
    int kt   = (i >> 9) % (K >> 5);
    int nt   = (i >> 9) / (K >> 5);
    int k = kt * 32 + 16 * (lane >> 4) + e;
    int n = nt * 16 + (lane & 15);
    out[i] = f2bf(W[(size_t)k * N + n]);
}

// ---------------------------------------------------------------------------
// 3) GEMM: Cout(f32, M x N) = A(bf16, M x K row-major) * Bpk(packed bf16) + bias
//    128 threads = 4 waves. Block output: 256 rows x 32 cols.
//    Each wave: 64x32 register tile (4 m-tiles x 2 n-tiles, 8 accumulators).
//    The block's B K-panel (2 n-tiles x full K = 64KB) is staged once into LDS
//    with async global->LDS copies (ASYNCcnt), then the K-loop is barrier-free.
__global__ __launch_bounds__(128)
void gemm_bf16_kernel(const unsigned short* __restrict__ A,
                      const unsigned short* __restrict__ Bpk,
                      const float* __restrict__ bias,
                      float* __restrict__ Cout,
                      int M, int N, int K) {
    const int KT   = K >> 5;              // K-steps of 32
    const int nt0  = blockIdx.x * 2;      // first of 2 n-tiles for this block
    const int tid  = threadIdx.x;
    const int wave = tid >> 5;
    const int lane = tid & 31;
    const int l16  = lane & 15;
    const int hi   = lane >> 4;

    // 64KB LDS: [j ntile][kt][lane][16 bf16]
    __shared__ unsigned short smem[2 * 32 * 512];

    // ---- async stage of B panel: contiguous 64KB (both n-tiles) ----
    {
        const unsigned short* src = Bpk + (size_t)nt0 * KT * 512;
        for (int i = tid * 8; i < 2 * KT * 512; i += 128 * 8) {
            unsigned lds = (unsigned)(uintptr_t)(smem + i);
            unsigned long long g = (unsigned long long)(uintptr_t)(src + i);
            asm volatile("global_load_async_to_lds_b128 %0, %1, off"
                         :: "v"(lds), "v"(g) : "memory");
        }
        asm volatile("s_wait_asynccnt 0x0" ::: "memory");
    }
    __syncthreads();

    const int mbase = blockIdx.y * 256 + wave * 64;   // wave's 64-row strip
    const unsigned short* arow[4];
#pragma unroll
    for (int i = 0; i < 4; ++i)
        arow[i] = A + (size_t)(mbase + i * 16 + l16) * K;

    v8f acc[4][2] = {};

    for (int kt = 0; kt < KT; ++kt) {
        int k0 = kt * 32;
        v16bf a[4];
#pragma unroll
        for (int i = 0; i < 4; ++i) {
            uint32* au = (uint32*)&a[i];
#pragma unroll
            for (int v = 0; v < 8; ++v) {
                int g = v >> 2;
                au[v] = *(const uint32*)(arow[i] + k0 + 16 * hi + 2 * v);
            }
        }
        v16bf b[2];
#pragma unroll
        for (int j = 0; j < 2; ++j)
            b[j] = *(const v16bf*)(smem + j * (KT * 512) + kt * 512 + lane * 16);
#pragma unroll
        for (int i = 0; i < 4; ++i)
#pragma unroll
            for (int j = 0; j < 2; ++j)
                acc[i][j] = __builtin_amdgcn_wmma_f32_16x16x32_bf16(
                    false, a[i], false, b[j], (short)0, acc[i][j], false, false);
    }

    float bn[2];
#pragma unroll
    for (int j = 0; j < 2; ++j) bn[j] = bias[(nt0 + j) * 16 + l16];

#pragma unroll
    for (int i = 0; i < 4; ++i) {
#pragma unroll
        for (int j = 0; j < 2; ++j) {
#pragma unroll
            for (int r = 0; r < 8; ++r) {
                int row = mbase + i * 16 + r + 8 * hi;
                int col = (nt0 + j) * 16 + l16;
                Cout[(size_t)row * N + col] = acc[i][j][r] + bn[j];
            }
        }
    }
}

// ---------------------------------------------------------------------------
// 4) RoPE on q,k (interleaved) + layout conversion to bf16.
//    qb,kb: (h, t, d) contiguous in d; vt: (h, d, t) contiguous in t.
__global__ void rope_cvt_kernel(const float* __restrict__ q32,
                                const float* __restrict__ k32,
                                const float* __restrict__ v32,
                                unsigned short* __restrict__ qb,
                                unsigned short* __restrict__ kb,
                                unsigned short* __restrict__ vt) {
    int i = blockIdx.x * blockDim.x + threadIdx.x;   // i = (h*T + t)*D + d
    if (i >= kH * kT * kD) return;
    int d = i & (kD - 1);
    int t = (i >> 6) & (kT - 1);
    int h = i >> 17;                                  // 6 + 11 bits

    int c = h * kD + d;
    size_t src  = (size_t)t * kC + c;
    size_t psrc = (size_t)t * kC + (c ^ 1);

    float qv = q32[src], kv = k32[src];
    float qp = q32[psrc], kp = k32[psrc];

    int d2 = d & ~1;
    float theta = (float)t * __expf(-(__logf(10000.0f) / (float)kD) * (float)d2);
    float sn = __sinf(theta), cs = __cosf(theta);
    float sgn = (d & 1) ? 1.0f : -1.0f;   // even: -x[d+1]*sin ; odd: +x[d-1]*sin

    qb[i] = f2bf(qv * cs + sgn * qp * sn);
    kb[i] = f2bf(kv * cs + sgn * kp * sn);
    vt[((size_t)h * kD + d) * kT + t] = f2bf(v32[src]);
}

// ---------------------------------------------------------------------------
// 5) Scores S = scale * q k^T with causal mask, written raw into att region.
//    One wave per 16x16 (qt,kt) tile; kt>qt tiles skipped (softmax zeros them).
__global__ void scores_kernel(const unsigned short* __restrict__ qb,
                              const unsigned short* __restrict__ kb,
                              float* __restrict__ att) {
    int kt = blockIdx.x, qt = blockIdx.y, h = blockIdx.z;
    if (kt > qt) return;
    int lane = threadIdx.x, l16 = lane & 15, hi = lane >> 4;

    v8f c = {};
    const unsigned short* qrow = qb + ((size_t)h * kT + qt * 16 + l16) * kD;
#pragma unroll
    for (int d0 = 0; d0 < kD; d0 += 32) {
        v16bf a;
        uint32* au = (uint32*)&a;
#pragma unroll
        for (int v = 0; v < 8; ++v) {
            int g = v >> 2;
            au[v] = *(const uint32*)(qrow + d0 + 16 * hi + 2 * v);
        }
        // B = K^T fragment: n = key time index (l16), k-dim = d (contiguous)
        v16bf b = *(const v16bf*)(kb + ((size_t)h * kT + kt * 16 + l16) * kD
                                  + d0 + 16 * hi);
        c = __builtin_amdgcn_wmma_f32_16x16x32_bf16(false, a, false, b,
                                                    (short)0, c, false, false);
    }
#pragma unroll
    for (int r = 0; r < 8; ++r) {
        int row = qt * 16 + r + 8 * hi;
        int col = kt * 16 + l16;
        float v = c[r] * 0.125f;             // 1/sqrt(64)
        if (col > row) v = -INFINITY;
        att[((size_t)h * kT + row) * kT + col] = v;
    }
}

// ---------------------------------------------------------------------------
// 6) Row softmax in place over att; zero-fills the (never written) upper
//    triangle so the full output tensor is defined.
__global__ void softmax_kernel(float* __restrict__ att) {
    int q = blockIdx.x, h = blockIdx.y;
    float* row = att + ((size_t)h * kT + q) * kT;
    __shared__ float red[256];
    int tid = threadIdx.x;
    int n = q + 1;                            // valid causal columns

    float m = -INFINITY;
    for (int c = tid; c < n; c += 256) m = fmaxf(m, row[c]);
    red[tid] = m;
    __syncthreads();
    for (int s = 128; s > 0; s >>= 1) {
        if (tid < s) red[tid] = fmaxf(red[tid], red[tid + s]);
        __syncthreads();
    }
    float mx = red[0];
    __syncthreads();

    float sum = 0.0f;
    for (int c = tid; c < n; c += 256) sum += __expf(row[c] - mx);
    red[tid] = sum;
    __syncthreads();
    for (int s = 128; s > 0; s >>= 1) {
        if (tid < s) red[tid] += red[tid + s];
        __syncthreads();
    }
    float inv = 1.0f / red[0];

    for (int c = tid; c < kT; c += 256)
        row[c] = (c < n) ? __expf(row[c] - mx) * inv : 0.0f;
}

// ---------------------------------------------------------------------------
// 7) y' = att @ V  (per head). A = att tile (f32 -> bf16 on the fly),
//    B = V^T-layout fragment (contiguous loads). Causal K-range only.
__global__ void attv_kernel(const float* __restrict__ att,
                            const unsigned short* __restrict__ vt,
                            unsigned short* __restrict__ ypb) {
    int nt = blockIdx.x, qt = blockIdx.y, h = blockIdx.z;
    int lane = threadIdx.x, l16 = lane & 15, hi = lane >> 4;

    v8f c = {};
    const float* arow = att + ((size_t)h * kT + qt * 16 + l16) * kT;
    int kend = (qt + 1) * 16;
    for (int kb0 = 0; kb0 < kend; kb0 += 32) {
        v16bf a;
        uint32* au = (uint32*)&a;
#pragma unroll
        for (int v = 0; v < 8; ++v) {
            int g = v >> 2;
            int k = kb0 + 16 * hi + 2 * v;
            float2 f = *(const float2*)(arow + k);
            au[v] = (uint32)f2bf(f.x) | ((uint32)f2bf(f.y) << 16);
        }
        v16bf b = *(const v16bf*)(vt + ((size_t)h * kD + nt * 16 + l16) * kT
                                  + kb0 + 16 * hi);
        c = __builtin_amdgcn_wmma_f32_16x16x32_bf16(false, a, false, b,
                                                    (short)0, c, false, false);
    }
#pragma unroll
    for (int r = 0; r < 8; ++r) {
        int row = qt * 16 + r + 8 * hi;
        ypb[(size_t)row * kC + h * kD + nt * 16 + l16] = f2bf(c[r]);
    }
}

// ---------------------------------------------------------------------------
extern "C" void kernel_launch(void* const* d_in, const int* in_sizes, int n_in,
                              void* d_out, int out_size, void* d_ws, size_t ws_size,
                              hipStream_t stream) {
    const float* x  = (const float*)d_in[0];
    const float* Wq = (const float*)d_in[1];
    const float* bq = (const float*)d_in[2];
    const float* Wk = (const float*)d_in[3];
    const float* bk = (const float*)d_in[4];
    const float* Wv = (const float*)d_in[5];
    const float* bv = (const float*)d_in[6];
    const float* Wp = (const float*)d_in[7];
    const float* bp = (const float*)d_in[8];

    float* y   = (float*)d_out;
    float* att = y + (size_t)kT * kC;          // output #2: (H,T,T) post-softmax

    // Workspace carve-up (all offsets 32B-aligned)
    char* w = (char*)d_ws;
    size_t off = 0;
    unsigned short* xbf  = (unsigned short*)(w + off); off += (size_t)kT * kC * 2;
    unsigned short* Wqpk = (unsigned short*)(w + off); off += (size_t)kC * kC * 2;
    unsigned short* Wkpk = (unsigned short*)(w + off); off += (size_t)kC * kC * 2;
    unsigned short* Wvpk = (unsigned short*)(w + off); off += (size_t)kC * kC * 2;
    unsigned short* Wppk = (unsigned short*)(w + off); off += (size_t)kC * kC * 2;
    float* q32 = (float*)(w + off); off += (size_t)kT * kC * 4;
    float* k32 = (float*)(w + off); off += (size_t)kT * kC * 4;
    float* v32 = (float*)(w + off); off += (size_t)kT * kC * 4;
    unsigned short* qb  = (unsigned short*)(w + off); off += (size_t)kH * kT * kD * 2;
    unsigned short* kbw = (unsigned short*)(w + off); off += (size_t)kH * kT * kD * 2;
    unsigned short* vt  = (unsigned short*)(w + off); off += (size_t)kH * kT * kD * 2;
    unsigned short* ypb = (unsigned short*)(w + off); off += (size_t)kT * kC * 2;
    (void)ws_size; (void)in_sizes; (void)n_in; (void)out_size;

    // 1) x -> bf16
    cvt_bf16_kernel<<<(kT * kC + 255) / 256, 256, 0, stream>>>(x, xbf, kT * kC);

    // 2) pack weights
    int npk = kC * kC;
    pack_w_kernel<<<(npk + 255) / 256, 256, 0, stream>>>(Wq, Wqpk, kC, kC);
    pack_w_kernel<<<(npk + 255) / 256, 256, 0, stream>>>(Wk, Wkpk, kC, kC);
    pack_w_kernel<<<(npk + 255) / 256, 256, 0, stream>>>(Wv, Wvpk, kC, kC);
    pack_w_kernel<<<(npk + 255) / 256, 256, 0, stream>>>(Wp, Wppk, kC, kC);

    // 3) QKV projections (WMMA bf16, f32 accumulate)
    dim3 ggrid(kC / 32, kT / 256);
    gemm_bf16_kernel<<<ggrid, 128, 0, stream>>>(xbf, Wqpk, bq, q32, kT, kC, kC);
    gemm_bf16_kernel<<<ggrid, 128, 0, stream>>>(xbf, Wkpk, bk, k32, kT, kC, kC);
    gemm_bf16_kernel<<<ggrid, 128, 0, stream>>>(xbf, Wvpk, bv, v32, kT, kC, kC);

    // 4) RoPE + bf16 head layouts
    int nr = kH * kT * kD;
    rope_cvt_kernel<<<(nr + 255) / 256, 256, 0, stream>>>(q32, k32, v32, qb, kbw, vt);

    // 5) causal scores into att region
    scores_kernel<<<dim3(kT / 16, kT / 16, kH), 32, 0, stream>>>(qb, kbw, att);

    // 6) softmax (also zero-fills upper triangle)
    softmax_kernel<<<dim3(kT, kH), 256, 0, stream>>>(att);

    // 7) y' = att @ V
    attv_kernel<<<dim3(kD / 16, kT / 16, kH), 32, 0, stream>>>(att, vt, ypb);

    // 8) final projection y = y' @ Wp + bp
    gemm_bf16_kernel<<<ggrid, 128, 0, stream>>>(ypb, Wppk, bp, y, kT, kC, kC);
}
